// SemiSinkhornKnopp_75411035783868
// MI455X (gfx1250) — matrix operand order repaired and checked by reading, not compiled, not run on hardware
//
#include <hip/hip_runtime.h>

// ---------------------------------------------------------------------------
// Semi-relaxed Sinkhorn-Knopp on MI455X (gfx1250, wave32).
//  * Q cached in bf16 (128 MB -> L2-resident on MI455X's 192 MB L2)
//  * row-GEMV done with V_WMMA_F32_16X16X32_BF16 (b broadcast across B cols)
//  * fused iteration: Q@b -> a_new -> partial Q^T a_new column sums
//  * fixed 1000-launch schedule, data-dependent early exit via ws flag
// ---------------------------------------------------------------------------

#define KDIM     1024
#define FI       (1.0f / 1.1f)       // gamma/(gamma+eps)
#define STOPERR2 1e-12f              // (1e-6)^2
#define MAXIT    1000

typedef __attribute__((ext_vector_type(16))) __bf16    v16bf;
typedef __attribute__((ext_vector_type(8)))  float     v8f;
typedef __attribute__((ext_vector_type(4)))  unsigned  v4u;
typedef __attribute__((ext_vector_type(8)))  unsigned  v8u;

__device__ __forceinline__ float bf2f(unsigned short u) {
    return __uint_as_float(((unsigned)u) << 16);
}
__device__ __forceinline__ unsigned short f2bf(float f) {
    __bf16 h = (__bf16)f;                      // RNE convert
    return __builtin_bit_cast(unsigned short, h);
}
__device__ __forceinline__ float wave_max(float x) {
    #pragma unroll
    for (int o = 16; o > 0; o >>= 1) x = fmaxf(x, __shfl_xor(x, o, 32));
    return x;
}
__device__ __forceinline__ float wave_sum(float x) {
    #pragma unroll
    for (int o = 16; o > 0; o >>= 1) x += __shfl_xor(x, o, 32);
    return x;
}

// ---------------------------------------------------------------------------
// init: b = 1/K, colsum = wsum = 0, a = 1, flag = 0, loss accumulator = 0
// ---------------------------------------------------------------------------
__global__ void sk_init_kernel(float* __restrict__ b, float* __restrict__ colsum,
                               float* __restrict__ wsum, float* __restrict__ a,
                               int* __restrict__ flag, float* __restrict__ loss,
                               int N) {
    int i = blockIdx.x * 256 + threadIdx.x;
    if (i < KDIM) { b[i] = 1.0f / (float)KDIM; colsum[i] = 0.0f; wsum[i] = 0.0f; }
    if (i < N) a[i] = 1.0f;
    if (i == 0) { *flag = 0; *loss = 0.0f; }
}

// ---------------------------------------------------------------------------
// Pc/Q precompute: one wave per row.  x = 10*P;  Q = exp(10*(x - lse(x)))
// stored bf16 (row-major).  Shuffle-only reductions, no barriers.
// ---------------------------------------------------------------------------
__global__ void __launch_bounds__(256) sk_pcq_kernel(const float* __restrict__ P,
                                                     unsigned short* __restrict__ Qb,
                                                     int N) {
    const int wave = threadIdx.x >> 5;
    const int lane = threadIdx.x & 31;
    const int row  = blockIdx.x * 8 + wave;
    if (row >= N) return;
    const float* p = P + (size_t)row * KDIM;

    float x[32];
    float m = -3.402823466e38f;
    #pragma unroll
    for (int t = 0; t < 32; ++t) {
        x[t] = 10.0f * p[lane + 32 * t];
        m = fmaxf(m, x[t]);
    }
    m = wave_max(m);
    float s = 0.0f;
    #pragma unroll
    for (int t = 0; t < 32; ++t) s += __expf(x[t] - m);
    s = wave_sum(s);
    const float lse = m + __logf(s);

    unsigned short* q = Qb + (size_t)row * KDIM;
    #pragma unroll
    for (int t = 0; t < 32; ++t)
        q[lane + 32 * t] = f2bf(__expf(10.0f * (x[t] - lse)));
}

// ---------------------------------------------------------------------------
// Fused Sinkhorn iteration.  256 threads = 8 waves; each wave owns 16 rows;
// block covers 128 rows; grid = N/128.
//   phase 1 (WMMA): dot[m] = sum_k Q[m,k]*b[k] via v_wmma_f32_16x16x32_bf16,
//                   b broadcast across all 16 B-matrix columns.
//   phase 2: a_new[m] = (1/N)/dot[m]
//   phase 3: colsum[j] += sum_m a_new[m]*Q[m,j]  (LDS f32 atomics -> global)
// ---------------------------------------------------------------------------
__global__ void __launch_bounds__(256) sk_iter_kernel(
        const unsigned short* __restrict__ Qb, const float* __restrict__ b,
        float* __restrict__ a, float* __restrict__ colsum_g,
        const int* __restrict__ flag, int N) {
    if (*(volatile const int*)flag) return;

    __shared__ __align__(32) unsigned short b_lds[KDIM];
    __shared__ float cs_lds[KDIM];
    __shared__ float a_lds[128];          // 8 waves * 16 rows

    for (int j = threadIdx.x; j < KDIM; j += 256) {
        b_lds[j]  = f2bf(b[j]);
        cs_lds[j] = 0.0f;
    }
    __syncthreads();

    const int wave = threadIdx.x >> 5;
    const int lane = threadIdx.x & 31;
    const int g    = lane >> 4;           // lane group (K-halves split)
    const int mloc = lane & 15;           // M index inside the 16-row tile
    const int row0 = (blockIdx.x * 8 + wave) * 16;

    const unsigned short* qA = Qb + (size_t)(row0 + mloc) * KDIM;  // A frag rows
    const unsigned short* qT = Qb + (size_t)row0 * KDIM;           // tile base

    v8f acc = {0.f, 0.f, 0.f, 0.f, 0.f, 0.f, 0.f, 0.f};

    #pragma unroll 4
    for (int k0 = 0; k0 < KDIM; k0 += 32) {
        __builtin_prefetch(qA + k0 + 128, 0, 0);   // global_prefetch_b8
        // 16-bit A 16x32 layout: lane holds K runs [g*8,g*8+8) and [16+g*8,+8)
        v4u lo = *(const v4u*)(qA + k0 + g * 8);
        v4u hi = *(const v4u*)(qA + k0 + 16 + g * 8);
        v8u au;
        au[0] = lo[0]; au[1] = lo[1]; au[2] = lo[2]; au[3] = lo[3];
        au[4] = hi[0]; au[5] = hi[1]; au[6] = hi[2]; au[7] = hi[3];
        v16bf A = __builtin_bit_cast(v16bf, au);
        // 16-bit B 32x16 layout: lane group g holds K runs [g*8,g*8+8) and [16+g*8,+8), broadcast
        // the same b slice to every N column -> all D columns hold the dot.
        v4u blo = *(const v4u*)(b_lds + k0 + g * 8);
        v4u bhi = *(const v4u*)(b_lds + k0 + 16 + g * 8);
        v8u bu;
        bu[0] = blo[0]; bu[1] = blo[1]; bu[2] = blo[2]; bu[3] = blo[3];
        bu[4] = bhi[0]; bu[5] = bhi[1]; bu[6] = bhi[2]; bu[7] = bhi[3];
        v16bf B = __builtin_bit_cast(v16bf, bu);
        acc = __builtin_amdgcn_wmma_f32_16x16x32_bf16(
                  false, A, false, B, (short)0, acc, false, false);
    }

    // D layout: lanes 0-15 hold M=r in VGPR r; lanes 16-31 hold M=8+r.
    if (lane == 0) {
        #pragma unroll
        for (int r = 0; r < 8; ++r) a_lds[wave * 16 + r] = acc[r];
    }
    if (lane == 16) {
        #pragma unroll
        for (int r = 0; r < 8; ++r) a_lds[wave * 16 + 8 + r] = acc[r];
    }
    __syncthreads();

    if (lane < 16) {
        const float dot = a_lds[wave * 16 + lane];
        const float an  = (1.0f / (float)N) / dot;
        a[row0 + lane] = an;              // needed by the final OT pass
        a_lds[wave * 16 + lane] = an;
    }
    __syncthreads();

    float am[16];
    #pragma unroll
    for (int m = 0; m < 16; ++m) am[m] = a_lds[wave * 16 + m];

    for (int j = lane; j < KDIM; j += 32) {
        float s = 0.0f;
        #pragma unroll
        for (int m = 0; m < 16; ++m)
            s = fmaf(am[m], bf2f(qT[(size_t)m * KDIM + j]), s);
        atomicAdd(&cs_lds[j], s);         // ds_add_f32
    }
    __syncthreads();

    for (int j = threadIdx.x; j < KDIM; j += 256)
        atomicAdd(&colsum_g[j], cs_lds[j]);
}

// ---------------------------------------------------------------------------
// b update + convergence test.  One block of K threads.
//   b_new = (Pb / colsum)^fi ; err = ||b_new - b|| ; reset colsum.
// ---------------------------------------------------------------------------
__global__ void __launch_bounds__(KDIM) sk_bupdate_kernel(
        float* __restrict__ b, float* __restrict__ colsum_g, int* __restrict__ flag) {
    if (*(volatile const int*)flag) return;
    __shared__ float red[32];

    const int j = threadIdx.x;
    const float c = colsum_g[j];
    colsum_g[j] = 0.0f;                               // ready for next iter
    const float bn = __powf((1.0f / (float)KDIM) / c, FI);
    const float d  = bn - b[j];
    b[j] = bn;

    float e2 = wave_sum(d * d);
    const int wave = j >> 5, lane = j & 31;
    if (lane == 0) red[wave] = e2;
    __syncthreads();
    if (wave == 0) {
        float v = wave_sum(red[lane]);
        if (lane == 0 && v <= STOPERR2) *flag = 1;
    }
}

// ---------------------------------------------------------------------------
// Final pass: OT_plan = N*a*Q*b^T, loss = mean_i sum_j OT*Pc (Pc recomputed
// exactly from P), column sums for w.  One wave per row, 8 rows per wave.
// ---------------------------------------------------------------------------
__global__ void __launch_bounds__(256) sk_final_kernel(
        const float* __restrict__ P, const unsigned short* __restrict__ Qb,
        const float* __restrict__ a, const float* __restrict__ b,
        float* __restrict__ out, float* __restrict__ loss_acc,
        float* __restrict__ wsum, int N) {
    __shared__ float w_lds[KDIM];
    for (int j = threadIdx.x; j < KDIM; j += 256) w_lds[j] = 0.0f;
    __syncthreads();

    const int wave = threadIdx.x >> 5;
    const int lane = threadIdx.x & 31;
    const int row0 = (blockIdx.x * 8 + wave) * 8;     // 8 rows per wave

    float bj[32];
    #pragma unroll
    for (int t = 0; t < 32; ++t) bj[t] = b[lane + 32 * t];
    float wacc[32];
    #pragma unroll
    for (int t = 0; t < 32; ++t) wacc[t] = 0.0f;
    float lacc = 0.0f;

    for (int r = 0; r < 8; ++r) {
        const int row = row0 + r;
        const float* p          = P  + (size_t)row * KDIM;
        const unsigned short* q = Qb + (size_t)row * KDIM;
        float x[32];
        float m = -3.402823466e38f;
        #pragma unroll
        for (int t = 0; t < 32; ++t) {
            x[t] = 10.0f * p[lane + 32 * t];
            m = fmaxf(m, x[t]);
        }
        m = wave_max(m);
        float s = 0.0f;
        #pragma unroll
        for (int t = 0; t < 32; ++t) s += __expf(x[t] - m);
        s = wave_sum(s);
        const float lse = m + __logf(s);
        const float ai  = a[row];
        float* o = out + (size_t)row * KDIM;
        #pragma unroll
        for (int t = 0; t < 32; ++t) {
            const float pc = lse - x[t];
            const float ot = (float)N * ai * bf2f(q[lane + 32 * t]) * bj[t];
            o[lane + 32 * t] = ot;
            lacc = fmaf(ot, pc, lacc);
            wacc[t] += ot;
        }
    }

    lacc = wave_sum(lacc);
    if (lane == 0) atomicAdd(loss_acc, lacc / (float)N);
    #pragma unroll
    for (int t = 0; t < 32; ++t) atomicAdd(&w_lds[lane + 32 * t], wacc[t]);
    __syncthreads();
    for (int j = threadIdx.x; j < KDIM; j += 256)
        atomicAdd(&wsum[j], w_lds[j]);
}

// ---------------------------------------------------------------------------
// reg = sum_j tgt*(log tgt - log(w_j + 1e-7)),  w = wsum/N, tgt = 1/K
// ---------------------------------------------------------------------------
__global__ void __launch_bounds__(KDIM) sk_reg_kernel(
        const float* __restrict__ wsum, float* __restrict__ reg_out, int N) {
    __shared__ float red[32];
    const int j = threadIdx.x;
    const float w   = wsum[j] / (float)N;
    const float tgt = 1.0f / (float)KDIM;
    float v = tgt * (__logf(tgt) - __logf(w + 1e-7f));
    v = wave_sum(v);
    const int wave = j >> 5, lane = j & 31;
    if (lane == 0) red[wave] = v;
    __syncthreads();
    if (wave == 0) {
        float t = wave_sum(red[lane]);
        if (lane == 0) *reg_out = t;
    }
}

// ---------------------------------------------------------------------------
extern "C" void kernel_launch(void* const* d_in, const int* in_sizes, int n_in,
                              void* d_out, int out_size, void* d_ws, size_t ws_size,
                              hipStream_t stream) {
    const float* P = (const float*)d_in[0];
    const int N = in_sizes[0] / KDIM;            // 65536

    float* out      = (float*)d_out;
    float* loss_acc = out + (size_t)N * KDIM;    // d_out[N*K]
    float* reg_out  = loss_acc + 1;              // d_out[N*K + 1]

    // workspace layout
    char* ws = (char*)d_ws;
    unsigned short* Qb = (unsigned short*)ws;    ws += (size_t)N * KDIM * sizeof(unsigned short);
    float* b       = (float*)ws;                 ws += KDIM * sizeof(float);
    float* colsum  = (float*)ws;                 ws += KDIM * sizeof(float);
    float* wsum    = (float*)ws;                 ws += KDIM * sizeof(float);
    float* a       = (float*)ws;                 ws += (size_t)N * sizeof(float);
    int*   flag    = (int*)ws;

    hipLaunchKernelGGL(sk_init_kernel, dim3((N + 255) / 256), dim3(256), 0, stream,
                       b, colsum, wsum, a, flag, loss_acc, N);
    hipLaunchKernelGGL(sk_pcq_kernel, dim3(N / 8), dim3(256), 0, stream, P, Qb, N);

    for (int it = 0; it < MAXIT; ++it) {
        hipLaunchKernelGGL(sk_iter_kernel, dim3(N / 128), dim3(256), 0, stream,
                           Qb, b, a, colsum, flag, N);
        hipLaunchKernelGGL(sk_bupdate_kernel, dim3(1), dim3(KDIM), 0, stream,
                           b, colsum, flag);
    }

    hipLaunchKernelGGL(sk_final_kernel, dim3(N / 64), dim3(256), 0, stream,
                       P, Qb, a, b, out, loss_acc, wsum, N);
    hipLaunchKernelGGL(sk_reg_kernel, dim3(1), dim3(KDIM), 0, stream,
                       wsum, reg_out, N);
}
